// DLRMModel_15745350107453
// MI455X (gfx1250) — matrix operand order repaired and checked by reading, not compiled, not run on hardware
//
#include <hip/hip_runtime.h>

typedef __attribute__((ext_vector_type(16))) _Float16 v16h;
typedef __attribute__((ext_vector_type(8)))  _Float16 v8h;
typedef __attribute__((ext_vector_type(8)))  float    v8f;

#define BATCH      16384
#define NUM_FIELDS 26
#define VOCAB      100000
#define BN_SCALE   0.99999500003749969f   // 1/sqrt(1+1e-5)

// ---------------------------------------------------------------------------
// WMMA fragment loads (wave32; A and B both use contiguous k = 16*(lane/16) + s)
// ---------------------------------------------------------------------------

// A-fragment 16x32 f16 from row-major activations, row stride lda (halves).
__device__ __forceinline__ v16h load_a16x32(const _Float16* a, int lda, int lane) {
  const _Float16* p = a + (size_t)(lane & 15) * lda + ((lane >> 4) << 4);
  union { v16h v; v8h h[2]; } u;
  u.h[0] = *(const v8h*)(p);       // K = hi16 + 0..7
  u.h[1] = *(const v8h*)(p + 8);   // K = hi16 + 8..15
  return u.v;
}

// B-fragment 32x16 f16 from K-major (transposed) weights, row stride ldk (halves).
__device__ __forceinline__ v16h load_b16x32(const _Float16* wt, int ldk, int lane) {
  const _Float16* p = wt + (size_t)(lane & 15) * ldk + ((lane >> 4) << 4);
  return *(const v16h*)(p);        // 16 contiguous K values
}

// Async global -> LDS copy of 16 bytes per lane (ASYNCcnt-tracked).
__device__ __forceinline__ void async_copy_b128(_Float16* lds_dst, const _Float16* src) {
  asm volatile("global_load_async_to_lds_b128 %0, %1, off"
               :: "v"((unsigned)(uintptr_t)lds_dst), "v"(src) : "memory");
}

// ---------------------------------------------------------------------------
// Block-cooperative GEMM: 8 waves x 16 rows = 128 batch rows per block.
// Weights pre-transposed [N][ldw] f16; 16-col B tiles staged in LDS via async
// copy and shared by all waves. A fragments register-cached (KT <= 16).
// EPI: 0 = bias+BN+relu -> f16, 1 = bias only -> f16,
//      2 = raw f32 partial store, 3 = partial+bias+BN+relu -> f16.
// ---------------------------------------------------------------------------
template<int KT, int EPI>
__device__ __forceinline__ void block_gemm(const _Float16* __restrict__ A, int lda,
                                           const _Float16* __restrict__ WT, int ldw,
                                           const float* __restrict__ bias, float scale,
                                           _Float16* __restrict__ Cf16,
                                           float* __restrict__ Cf32,
                                           int N, _Float16* ldsB) {
  constexpr int ldk = KT * 32;                 // halves per B-tile row
  const int lane = threadIdx.x & 31;
  const int wave = threadIdx.x >> 5;

  const _Float16* Arow = A + (size_t)wave * 16 * lda;
  v16h a[KT];
#pragma unroll
  for (int k = 0; k < KT; ++k) a[k] = load_a16x32(Arow + k * 32, lda, lane);

  const int n  = lane & 15;
  const int mo = (lane >> 4) << 3;
  const size_t rbase = (size_t)wave * 16 + mo;

  for (int nt = 0; nt < (N >> 4); ++nt) {
    // Stage B tile (16 rows x ldk halves) into LDS: 2*ldk 16B chunks total.
    constexpr int chunksPerRow = ldk / 8;
    for (int t = threadIdx.x; t < 2 * ldk; t += 256) {
      const int r  = t / chunksPerRow;
      const int cc = t - r * chunksPerRow;
      async_copy_b128(ldsB + r * ldk + cc * 8,
                      WT + (size_t)(nt * 16 + r) * ldw + cc * 8);
    }
    asm volatile("s_wait_asynccnt 0x0" ::: "memory");
    __syncthreads();

    v8f c = {};
#pragma unroll
    for (int k = 0; k < KT; ++k) {
      v16h bv = load_b16x32(ldsB + k * 32, ldk, lane);
      c = __builtin_amdgcn_wmma_f32_16x16x32_f16(false, a[k], false, bv,
                                                 (short)0, c, false, false);
    }

    const int col = nt * 16 + n;
    if constexpr (EPI == 2) {
#pragma unroll
      for (int r = 0; r < 8; ++r)
        Cf32[(rbase + r) * N + col] = c[r];
    } else if constexpr (EPI == 3) {
      const float bb = bias[col];
#pragma unroll
      for (int r = 0; r < 8; ++r) {
        float v = (c[r] + Cf32[(rbase + r) * N + col] + bb) * scale;
        Cf16[(rbase + r) * N + col] = (_Float16)fmaxf(v, 0.0f);
      }
    } else {
      const float bb = bias[col];
#pragma unroll
      for (int r = 0; r < 8; ++r) {
        float v = (c[r] + bb) * scale;
        if constexpr (EPI == 0) v = fmaxf(v, 0.0f);
        Cf16[(rbase + r) * N + col] = (_Float16)v;
      }
    }
    __syncthreads();   // protect ldsB before next tile overwrites it
  }
}

// ---------------------------------------------------------------------------
// Prep: f32 weights [K][N] -> f16 transposed+padded [N][Kp]
// ---------------------------------------------------------------------------
__global__ void __launch_bounds__(256) prep_wT_kernel(const float* __restrict__ w,
                                                      _Float16* __restrict__ wt,
                                                      int K, int N, int Kp) {
  const int i = blockIdx.x * 256 + threadIdx.x;
  if (i >= N * Kp) return;
  const int n = i / Kp, k = i - n * Kp;
  wt[i] = (k < K) ? (_Float16)w[(size_t)k * N + n] : (_Float16)0.0f;
}

// x_dense [B][13] f32 -> [B][32] f16 zero-padded
__global__ void __launch_bounds__(256) prep_dense_kernel(const float* __restrict__ xd,
                                                         _Float16* __restrict__ dx) {
  const int i = blockIdx.x * 256 + threadIdx.x;
  const int b = i >> 5, k = i & 31;
  dx[i] = (k < 13) ? (_Float16)xd[b * 13 + k] : (_Float16)0.0f;
}

// ---------------------------------------------------------------------------
// Bottom MLP: 13->512->256->64
// ---------------------------------------------------------------------------
__global__ void __launch_bounds__(256) bottom_mlp_kernel(
    const _Float16* __restrict__ dx,
    const _Float16* __restrict__ w0T, const float* __restrict__ b0,
    const _Float16* __restrict__ w1T, const float* __restrict__ b1,
    const _Float16* __restrict__ w2T, const float* __restrict__ b2,
    _Float16* __restrict__ h0, _Float16* __restrict__ h1,
    _Float16* __restrict__ dout) {
  __shared__ alignas(64) _Float16 ldsB[16 * 512];   // 16 KB, max KT=16 tile
  const size_t r0 = (size_t)blockIdx.x * 128;
  block_gemm<1, 0>(dx + r0 * 32, 32, w0T, 32, b0, BN_SCALE,
                   h0 + r0 * 512, nullptr, 512, ldsB);
  asm volatile("s_wait_storecnt 0x0" ::: "memory");
  block_gemm<16, 0>(h0 + r0 * 512, 512, w1T, 512, b1, BN_SCALE,
                    h1 + r0 * 256, nullptr, 256, ldsB);
  asm volatile("s_wait_storecnt 0x0" ::: "memory");
  block_gemm<8, 1>(h1 + r0 * 256, 256, w2T, 256, b2, 1.0f,
                   dout + r0 * 64, nullptr, 64, ldsB);
}

// ---------------------------------------------------------------------------
// Fused embedding gather + gram (WMMA) + upper-tri interactions + feat concat.
// One wave per sample; 32x64 f16 tile staged in LDS serves as both A and B^T.
// ---------------------------------------------------------------------------
__global__ void __launch_bounds__(256) interact_kernel(
    const int* __restrict__ xs, const float* __restrict__ emb,
    const _Float16* __restrict__ dout, _Float16* __restrict__ feats) {
  __shared__ alignas(64) _Float16 tiles[8][32 * 64];
  const int lane = threadIdx.x & 31;
  const int wave = threadIdx.x >> 5;
  const size_t s = (size_t)blockIdx.x * 8 + wave;
  _Float16* T = tiles[wave];

  for (int f = 0; f < NUM_FIELDS; ++f) {
    const long long idx = (long long)xs[s * NUM_FIELDS + f];
    const float2 v = ((const float2*)(emb + ((size_t)f * VOCAB + idx) * 64))[lane];
    T[f * 64 + lane * 2 + 0] = (_Float16)v.x;
    T[f * 64 + lane * 2 + 1] = (_Float16)v.y;
  }
  T[26 * 64 + 2 * lane + 0] = dout[s * 64 + 2 * lane + 0];
  T[26 * 64 + 2 * lane + 1] = dout[s * 64 + 2 * lane + 1];
#pragma unroll
  for (int r = 27; r < 32; ++r) {
    T[r * 64 + 2 * lane + 0] = (_Float16)0.0f;
    T[r * 64 + 2 * lane + 1] = (_Float16)0.0f;
  }
  asm volatile("s_wait_dscnt 0x0" ::: "memory");

  v8f c[2][2] = {};
#pragma unroll
  for (int kt = 0; kt < 2; ++kt) {
    const v16h a0 = load_a16x32(T + kt * 32,           64, lane);
    const v16h a1 = load_a16x32(T + 16 * 64 + kt * 32, 64, lane);
    const v16h b0 = load_b16x32(T + kt * 32,           64, lane);
    const v16h b1 = load_b16x32(T + 16 * 64 + kt * 32, 64, lane);
    c[0][0] = __builtin_amdgcn_wmma_f32_16x16x32_f16(false, a0, false, b0, (short)0, c[0][0], false, false);
    c[0][1] = __builtin_amdgcn_wmma_f32_16x16x32_f16(false, a0, false, b1, (short)0, c[0][1], false, false);
    c[1][0] = __builtin_amdgcn_wmma_f32_16x16x32_f16(false, a1, false, b0, (short)0, c[1][0], false, false);
    c[1][1] = __builtin_amdgcn_wmma_f32_16x16x32_f16(false, a1, false, b1, (short)0, c[1][1], false, false);
  }

  _Float16* F = feats + s * 448;
  const int nn = lane & 15;
  const int mo = (lane >> 4) << 3;
#pragma unroll
  for (int ti = 0; ti < 2; ++ti)
#pragma unroll
    for (int tj = 0; tj < 2; ++tj)
#pragma unroll
      for (int r = 0; r < 8; ++r) {
        const int i = ti * 16 + mo + r;
        const int j = tj * 16 + nn;
        if (i < j && j < 27)
          F[26 * i - ((i * (i - 1)) >> 1) + (j - i - 1)] = (_Float16)c[ti][tj][r];
      }
  F[351 + 2 * lane + 0] = dout[s * 64 + 2 * lane + 0];
  F[351 + 2 * lane + 1] = dout[s * 64 + 2 * lane + 1];
  for (int t = lane; t < 33; t += 32) F[415 + t] = (_Float16)0.0f;
}

// ---------------------------------------------------------------------------
// Top MLP: 416->1024->512->256 (layer 1 K-split into 2x K=512 passes)
// ---------------------------------------------------------------------------
__global__ void __launch_bounds__(256) top_mlp_kernel(
    const _Float16* __restrict__ feats,
    const _Float16* __restrict__ w0T, const float* __restrict__ b0,
    const _Float16* __restrict__ w1T, const float* __restrict__ b1,
    const _Float16* __restrict__ w2T, const float* __restrict__ b2,
    _Float16* __restrict__ t0, _Float16* __restrict__ t1,
    _Float16* __restrict__ t2, float* __restrict__ part) {
  __shared__ alignas(64) _Float16 ldsB[16 * 512];   // 16 KB, max KT=16 tile
  const size_t r0 = (size_t)blockIdx.x * 128;
  block_gemm<13, 0>(feats + r0 * 448, 448, w0T, 416, b0, BN_SCALE,
                    t0 + r0 * 1024, nullptr, 1024, ldsB);
  asm volatile("s_wait_storecnt 0x0" ::: "memory");
  block_gemm<16, 2>(t0 + r0 * 1024, 1024, w1T, 1024, nullptr, 0.0f,
                    nullptr, part + r0 * 512, 512, ldsB);
  asm volatile("s_wait_storecnt 0x0" ::: "memory");
  block_gemm<16, 3>(t0 + r0 * 1024 + 512, 1024, w1T + 512, 1024, b1, BN_SCALE,
                    t1 + r0 * 512, part + r0 * 512, 512, ldsB);
  asm volatile("s_wait_storecnt 0x0" ::: "memory");
  block_gemm<16, 0>(t1 + r0 * 512, 512, w2T, 512, b2, BN_SCALE,
                    t2 + r0 * 256, nullptr, 256, ldsB);
}

// Final: 256 -> 1 + sigmoid (one thread per sample; trivial FLOPs).
__global__ void __launch_bounds__(256) final_kernel(const _Float16* __restrict__ t2,
                                                    const float* __restrict__ w3,
                                                    const float* __restrict__ b3,
                                                    float* __restrict__ out) {
  const int i = blockIdx.x * 256 + threadIdx.x;
  const _Float16* r = t2 + (size_t)i * 256;
  float acc = 0.0f;
#pragma unroll 8
  for (int k = 0; k < 256; ++k) acc += (float)r[k] * w3[k];
  acc += b3[0];
  out[i] = 1.0f / (1.0f + __expf(-acc));
}

// ---------------------------------------------------------------------------
extern "C" void kernel_launch(void* const* d_in, const int* in_sizes, int n_in,
                              void* d_out, int out_size, void* d_ws, size_t ws_size,
                              hipStream_t stream) {
  (void)in_sizes; (void)n_in; (void)out_size; (void)ws_size;
  const int*   xs  = (const int*)  d_in[0];
  const float* xd  = (const float*)d_in[1];
  const float* emb = (const float*)d_in[2];
  const float* bw0 = (const float*)d_in[3];  const float* bb0 = (const float*)d_in[4];
  const float* bw1 = (const float*)d_in[5];  const float* bb1 = (const float*)d_in[6];
  const float* bw2 = (const float*)d_in[7];  const float* bb2 = (const float*)d_in[8];
  const float* tw0 = (const float*)d_in[9];  const float* tb0 = (const float*)d_in[10];
  const float* tw1 = (const float*)d_in[11]; const float* tb1 = (const float*)d_in[12];
  const float* tw2 = (const float*)d_in[13]; const float* tb2 = (const float*)d_in[14];
  const float* tw3 = (const float*)d_in[15]; const float* tb3 = (const float*)d_in[16];
  float* out = (float*)d_out;

  char* base = (char*)d_ws;
  size_t off = 0;
  auto alloc = [&](size_t bytes) {
    void* p = base + off;
    off = (off + bytes + 255) & ~(size_t)255;
    return p;
  };
  _Float16* w0T = (_Float16*)alloc((size_t)512 * 32 * 2);
  _Float16* w1T = (_Float16*)alloc((size_t)256 * 512 * 2);
  _Float16* w2T = (_Float16*)alloc((size_t)64 * 256 * 2);
  _Float16* u0T = (_Float16*)alloc((size_t)1024 * 416 * 2);
  _Float16* u1T = (_Float16*)alloc((size_t)512 * 1024 * 2);
  _Float16* u2T = (_Float16*)alloc((size_t)256 * 512 * 2);
  _Float16* dx  = (_Float16*)alloc((size_t)BATCH * 32 * 2);
  _Float16* h0  = (_Float16*)alloc((size_t)BATCH * 512 * 2);
  _Float16* h1  = (_Float16*)alloc((size_t)BATCH * 256 * 2);
  _Float16* dof = (_Float16*)alloc((size_t)BATCH * 64 * 2);
  _Float16* fe  = (_Float16*)alloc((size_t)BATCH * 448 * 2);
  _Float16* t0  = (_Float16*)alloc((size_t)BATCH * 1024 * 2);
  float*    part= (float*)   alloc((size_t)BATCH * 512 * 4);
  _Float16* t1  = h0;  // reuse: bottom layer-0 act dead by then
  _Float16* t2  = h1;  // reuse: bottom layer-1 act dead by then

  prep_wT_kernel<<<(512 * 32   + 255) / 256, 256, 0, stream>>>(bw0, w0T, 13,   512,  32);
  prep_wT_kernel<<<(256 * 512  + 255) / 256, 256, 0, stream>>>(bw1, w1T, 512,  256,  512);
  prep_wT_kernel<<<(64  * 256  + 255) / 256, 256, 0, stream>>>(bw2, w2T, 256,  64,   256);
  prep_wT_kernel<<<(1024 * 416 + 255) / 256, 256, 0, stream>>>(tw0, u0T, 415,  1024, 416);
  prep_wT_kernel<<<(512 * 1024 + 255) / 256, 256, 0, stream>>>(tw1, u1T, 1024, 512,  1024);
  prep_wT_kernel<<<(256 * 512  + 255) / 256, 256, 0, stream>>>(tw2, u2T, 512,  256,  512);
  prep_dense_kernel<<<BATCH * 32 / 256, 256, 0, stream>>>(xd, dx);

  bottom_mlp_kernel<<<BATCH / 128, 256, 0, stream>>>(dx, w0T, bb0, w1T, bb1, w2T, bb2,
                                                     h0, h1, dof);
  interact_kernel<<<BATCH / 8, 256, 0, stream>>>(xs, emb, dof, fe);
  top_mlp_kernel<<<BATCH / 128, 256, 0, stream>>>(fe, u0T, tb0, u1T, tb1, u2T, tb2,
                                                  t0, t1, t2, part);
  final_kernel<<<BATCH / 256, 256, 0, stream>>>(t2, tw3, tb3, out);
}
